// FAE_FeatGraphConv_6107443495307
// MI455X (gfx1250) — matrix operand order repaired and checked by reading, not compiled, not run on hardware
//
#include <hip/hip_runtime.h>

typedef __attribute__((ext_vector_type(16))) __bf16 v16bf;
typedef __attribute__((ext_vector_type(8)))  float  v8f;

// ---------------------------------------------------------------------------
// Weight prep: W[K,N] f32 -> transposed bf16 split pair WT{hi,lo}[N,K].
// Transposed layout makes each WMMA B-fragment (16 consecutive K for one
// column) a single contiguous 32-byte load.
// ---------------------------------------------------------------------------
__global__ void prep_w(const float* __restrict__ W, __bf16* __restrict__ WThi,
                       __bf16* __restrict__ WTlo, int K, int N) {
  int i = blockIdx.x * blockDim.x + threadIdx.x;
  if (i >= K * N) return;
  int k = i / N, c = i % N;
  float w = W[i];
  __bf16 hv = (__bf16)w;
  WThi[(size_t)c * K + k] = hv;
  WTlo[(size_t)c * K + k] = (__bf16)(w - (float)hv);
}

// ---------------------------------------------------------------------------
// GEMM: Out[nrows,N] = concat(Xa[:,0:KA], Xb[:,0:KB]) @ W[KA+KB,N] + bias
// Split-bf16 WMMA (hi*hi + hi*lo + lo*hi) for fp32-class accuracy.
// One wave computes a 16-row x N stripe. Requires nrows % 16 == 0.
// Weights come pre-converted/transposed: WT{hi,lo} is [N][K] bf16.
// ---------------------------------------------------------------------------
template<int KA, int KB, int N, bool RELU>
__global__ __launch_bounds__(256)
void gemm_wmma(const float* __restrict__ Xa, const float* __restrict__ Xb,
               const __bf16* __restrict__ WThi, const __bf16* __restrict__ WTlo,
               const float* __restrict__ bias,
               float* __restrict__ Out, int nrows) {
  constexpr int K   = KA + KB;
  constexpr int NKB = K / 32;   // 32-wide K blocks
  constexpr int NNT = N / 16;   // 16-wide N tiles
  const int lane  = threadIdx.x & 31;
  const int wave  = threadIdx.x >> 5;
  const int m0    = (blockIdx.x * (blockDim.x >> 5) + wave) * 16;
  if (m0 >= nrows) return;                       // whole-wave exit only
  const int row   = m0 + (lane & 15);
  const int kb8   = (lane >> 4) * 8;             // unused; A also uses kb16
  const int col16 = lane & 15;                   // B/D column within tile
  const int kb16  = (lane >> 4) * 16;            // B half-wave K offset

  v8f acc[NNT];
  #pragma unroll
  for (int t = 0; t < NNT; ++t) acc[t] = v8f{};

  #pragma unroll
  for (int kblk = 0; kblk < NKB; ++kblk) {
    const int K0 = kblk * 32;
    // ---- A fragment: elems 0..7 -> K0+kb16.., elems 8..15 -> K0+kb16+8..
    float a[16];
    #pragma unroll
    for (int h = 0; h < 2; ++h) {
      const int kk = K0 + kb16 + h * 8;          // 8-aligned, never straddles Xa/Xb
      const float* srcp; int kcol, ld;
      if (KB == 0)        { srcp = Xa; kcol = kk;      ld = KA; }
      else if (kk < KA)   { srcp = Xa; kcol = kk;      ld = KA; }
      else                { srcp = Xb; kcol = kk - KA; ld = KB; }
      const float4* p = (const float4*)(srcp + (size_t)row * ld + kcol);
      float4 v0 = p[0], v1 = p[1];
      a[h*8+0]=v0.x; a[h*8+1]=v0.y; a[h*8+2]=v0.z; a[h*8+3]=v0.w;
      a[h*8+4]=v1.x; a[h*8+5]=v1.y; a[h*8+6]=v1.z; a[h*8+7]=v1.w;
    }
    v16bf ahi, alo;
    #pragma unroll
    for (int e = 0; e < 16; ++e) {
      __bf16 hv = (__bf16)a[e];
      ahi[e] = hv;
      alo[e] = (__bf16)(a[e] - (float)hv);
    }
    // ---- per N-tile: contiguous 32B B-fragment loads + 3 WMMAs
    #pragma unroll
    for (int t = 0; t < NNT; ++t) {
      const size_t boff = (size_t)(t * 16 + col16) * K + K0 + kb16;
      v16bf bhi = *(const v16bf*)(WThi + boff);
      v16bf blo = *(const v16bf*)(WTlo + boff);
      acc[t] = __builtin_amdgcn_wmma_f32_16x16x32_bf16(false, ahi, false, bhi,
                                                       (short)0, acc[t], false, false);
      acc[t] = __builtin_amdgcn_wmma_f32_16x16x32_bf16(false, ahi, false, blo,
                                                       (short)0, acc[t], false, false);
      acc[t] = __builtin_amdgcn_wmma_f32_16x16x32_bf16(false, alo, false, bhi,
                                                       (short)0, acc[t], false, false);
    }
  }
  // ---- epilogue: bias, relu, store (D: elem r -> row m0 + r + 8*(lane>>4))
  #pragma unroll
  for (int t = 0; t < NNT; ++t) {
    const int c = t * 16 + col16;
    const float b = bias[c];
    #pragma unroll
    for (int r = 0; r < 8; ++r) {
      const int m = m0 + r + 8 * (lane >> 4);
      float v = acc[t][r] + b;
      if (RELU) v = fmaxf(v, 0.0f);
      Out[(size_t)m * N + c] = v;
    }
  }
}

// agg = h (self-loop contribution), optionally cnt = 1
__global__ void init_agg(const float* __restrict__ h, float* __restrict__ agg,
                         float* __restrict__ cnt, int n, int c, int setcnt) {
  long long i = (long long)blockIdx.x * blockDim.x + threadIdx.x;
  long long tot = (long long)n * c;
  if (i < tot) agg[i] = h[i];
  if (setcnt && i < n) cnt[i] = 1.0f;
}

// edge-parallel scatter: thread = (edge, 4-channel chunk)
template<int C, bool COUNT>
__global__ void scatter_add(const float* __restrict__ h,
                            const int* __restrict__ src, const int* __restrict__ dst,
                            float* __restrict__ agg, float* __restrict__ cnt,
                            long long nwork) {
  long long tid = (long long)blockIdx.x * blockDim.x + threadIdx.x;
  if (tid >= nwork) return;
  constexpr int CH = C / 4;
  const int e  = (int)(tid / CH);
  const int ch = (int)(tid % CH);
  const int s = src[e], d = dst[e];
  const float4 v = *(const float4*)(h + (size_t)s * C + ch * 4);
  float* o = agg + (size_t)d * C + ch * 4;
  unsafeAtomicAdd(o + 0, v.x);
  unsafeAtomicAdd(o + 1, v.y);
  unsafeAtomicAdd(o + 2, v.z);
  unsafeAtomicAdd(o + 3, v.w);
  if (COUNT && ch == 0) unsafeAtomicAdd(cnt + d, 1.0f);
}

__global__ void mean_div(float* __restrict__ agg, const float* __restrict__ cnt,
                         int n, int c) {
  int i = blockIdx.x * blockDim.x + threadIdx.x;
  if (i < n * c) agg[i] /= cnt[i / c];
}

// out[i] = dot(h2o[i,0:32], Wl) + bl
__global__ void final_lin(const float* __restrict__ h, const float* __restrict__ Wl,
                          const float* __restrict__ bl, float* __restrict__ out, int n) {
  int i = blockIdx.x * blockDim.x + threadIdx.x;
  if (i >= n) return;
  float s = bl[0];
  #pragma unroll
  for (int c = 0; c < 32; ++c) s += h[(size_t)i * 32 + c] * Wl[c];
  out[i] = s;
}

extern "C" void kernel_launch(void* const* d_in, const int* in_sizes, int n_in,
                              void* d_out, int out_size, void* d_ws, size_t ws_size,
                              hipStream_t stream) {
  const float* x    = (const float*)d_in[0];
  const int*   ei   = (const int*)d_in[1];          // int32 (JAX x64 disabled)
  const float* W2_1 = (const float*)d_in[2];
  const float* b2_1 = (const float*)d_in[3];
  const float* W1_1 = (const float*)d_in[4];
  const float* b1_1 = (const float*)d_in[5];
  const float* W2_2 = (const float*)d_in[6];
  const float* b2_2 = (const float*)d_in[7];
  const float* W1_2 = (const float*)d_in[8];
  const float* b1_2 = (const float*)d_in[9];
  const float* Wl   = (const float*)d_in[10];
  const float* bl   = (const float*)d_in[11];

  const int N = in_sizes[0] / 128;                  // 100000
  const int E = in_sizes[1] / 2;                    // 1600000
  const int* src = ei;
  const int* dst = ei + E;

  float* ws   = (float*)d_ws;
  float* bufA = ws;                                 // N*64 : h1, then h2 (N*32)
  float* bufB = bufA + (size_t)N * 64;              // N*64 : agg1, then agg2
  float* bufC = bufB + (size_t)N * 64;              // N*64 : h1o, then h2o
  float* cnt  = bufC + (size_t)N * 64;              // N

  // bf16 weight area (64B-aligned past cnt): hi/lo transposed copies
  uintptr_t wbase = (uintptr_t)(cnt + N);
  wbase = (wbase + 63u) & ~(uintptr_t)63u;
  __bf16* w21hi = (__bf16*)wbase;                   // 128*64
  __bf16* w21lo = w21hi + 128 * 64;
  __bf16* w11hi = w21lo + 128 * 64;                 // 128*64
  __bf16* w11lo = w11hi + 128 * 64;
  __bf16* w22hi = w11lo + 128 * 64;                 // 64*32
  __bf16* w22lo = w22hi + 64 * 32;
  __bf16* w12hi = w22lo + 64 * 32;                  // 64*32
  __bf16* w12lo = w12hi + 64 * 32;

  // ---- weight prep (recomputed every call; deterministic) ----
  prep_w<<<(128 * 64 + 255) / 256, 256, 0, stream>>>(W2_1, w21hi, w21lo, 128, 64);
  prep_w<<<(128 * 64 + 255) / 256, 256, 0, stream>>>(W1_1, w11hi, w11lo, 128, 64);
  prep_w<<<(64 * 32 + 255) / 256, 256, 0, stream>>>(W2_2, w22hi, w22lo, 64, 32);
  prep_w<<<(64 * 32 + 255) / 256, 256, 0, stream>>>(W1_2, w12hi, w12lo, 64, 32);

  const int gemm_grid = (N + 127) / 128;            // 8 waves * 16 rows / block
  const dim3 blk(256);

  // ---- conv1 ----
  gemm_wmma<128, 0, 64, false><<<gemm_grid, blk, 0, stream>>>(x, nullptr, w21hi, w21lo, b2_1, bufA, N);
  long long tot1 = (long long)N * 64;
  init_agg<<<(int)((tot1 + 255) / 256), 256, 0, stream>>>(bufA, bufB, cnt, N, 64, 1);
  long long work1 = (long long)E * 16;
  scatter_add<64, true><<<(int)((work1 + 255) / 256), 256, 0, stream>>>(bufA, src, dst, bufB, cnt, work1);
  mean_div<<<(int)((tot1 + 255) / 256), 256, 0, stream>>>(bufB, cnt, N, 64);
  gemm_wmma<64, 64, 64, true><<<gemm_grid, blk, 0, stream>>>(bufA, bufB, w11hi, w11lo, b1_1, bufC, N);

  // ---- conv2 (reuse bufA/bufB; bufC holds h1o as input) ----
  gemm_wmma<64, 0, 32, false><<<gemm_grid, blk, 0, stream>>>(bufC, nullptr, w22hi, w22lo, b2_2, bufA, N);
  long long tot2 = (long long)N * 32;
  init_agg<<<(int)((tot2 + 255) / 256), 256, 0, stream>>>(bufA, bufB, cnt, N, 32, 0);
  long long work2 = (long long)E * 8;
  scatter_add<32, false><<<(int)((work2 + 255) / 256), 256, 0, stream>>>(bufA, src, dst, bufB, cnt, work2);
  mean_div<<<(int)((tot2 + 255) / 256), 256, 0, stream>>>(bufB, cnt, N, 32);
  gemm_wmma<32, 32, 32, true><<<gemm_grid, blk, 0, stream>>>(bufA, bufB, w12hi, w12lo, b1_2, bufC, N);

  // ---- final linear ----
  final_lin<<<(N + 255) / 256, 256, 0, stream>>>(bufC, Wl, bl, (float*)d_out, N);
}
